// BaselineModel_24026047054064
// MI455X (gfx1250) — matrix operand order repaired and checked by reading, not compiled, not run on hardware
//
#include <hip/hip_runtime.h>
#include <hip/hip_bf16.h>

// ---------------------------------------------------------------------------
// MI455X (gfx1250) EEG ResNet forward pass.
//  - conv1d as implicit GEMM on v_wmma_f32_16x16x32_f16 (wave32), f32 accum
//  - time-major LDS activations with zero halos -> branchless 32B B-operand loads
//  - weights pre-swizzled to WMMA A-operand lane order (f16, workspace)
//  - pooled activations staged via global_load_async_to_lds_b128 (ASYNCcnt)
// ---------------------------------------------------------------------------

typedef __attribute__((ext_vector_type(16))) _Float16 v16h;
typedef __attribute__((ext_vector_type(8)))  _Float16 v8h;
typedef __attribute__((ext_vector_type(8)))  float    v8f;

#define BATCH   32
#define NE      100
#define NTR     30
#define NT      1000
#define NSEL    25
#define T_STEM  500
#define T_POOL  250
#define T_DOWN  125
#define LT      4          // left time halo (rows) in every activation buffer

// time-major buffer geometry (halves)
#define IN_ROWS   1032     // stem input: ti in [-3, 1025] + LT=4  (Cin padded 32)
#define IN_CP     32
#define P_ROWS    264      // 250-length buffers: ti up to 258 (+4) < 264
#define P_CP      64
#define D_ROWS    136      // 125-length buffers: ti up to 130 (+4) < 136
#define D_CP      128

// ---------------------------------------------------------------------------
// JAX threefry2x32 (classic 20-round) — reproduces jax.random.key(42) stream.
// ---------------------------------------------------------------------------
__device__ __forceinline__ void tf2x32(unsigned k0, unsigned k1,
                                       unsigned c0, unsigned c1,
                                       unsigned& o0, unsigned& o1) {
  const unsigned ks2 = k0 ^ k1 ^ 0x1BD11BDAu;
  unsigned x0 = c0 + k0;
  unsigned x1 = c1 + k1;
  auto rnd = [&](int r) {
    x0 += x1;
    x1 = (x1 << r) | (x1 >> (32 - r));
    x1 ^= x0;
  };
  rnd(13); rnd(15); rnd(26); rnd(6);   x0 += k1;  x1 += ks2 + 1u;
  rnd(17); rnd(29); rnd(16); rnd(24);  x0 += ks2; x1 += k0 + 2u;
  rnd(13); rnd(15); rnd(26); rnd(6);   x0 += k0;  x1 += k1 + 3u;
  rnd(17); rnd(29); rnd(16); rnd(24);  x0 += k1;  x1 += ks2 + 4u;
  rnd(13); rnd(15); rnd(26); rnd(6);   x0 += ks2; x1 += k0 + 5u;
  o0 = x0; o1 = x1;
}

__device__ __forceinline__ float u01(unsigned bits) {
  union { unsigned u; float f; } cv;
  cv.u = (bits >> 9) | 0x3F800000u;
  return cv.f - 1.0f;
}

__device__ __forceinline__ float uniform_at(unsigned k0, unsigned k1, int p, int n) {
  const int h = n >> 1;
  unsigned o0, o1;
  if (p < h) { tf2x32(k0, k1, (unsigned)p, (unsigned)(p + h), o0, o1); return u01(o0); }
  tf2x32(k0, k1, (unsigned)(p - h), (unsigned)p, o0, o1);
  return u01(o1);
}

// ---------------------------------------------------------------------------
// Async global->LDS staging (CDNA5; tracked by ASYNCcnt).
// ---------------------------------------------------------------------------
__device__ __forceinline__ void async_ld_b128(unsigned lds_byte_off,
                                              const void* sbase, int vbyte_off) {
  asm volatile("global_load_async_to_lds_b128 %0, %1, %2"
               :: "v"(lds_byte_off), "v"(vbyte_off), "s"(sbase) : "memory");
}
__device__ __forceinline__ void wait_async0() {
  asm volatile("s_wait_asynccnt 0x0" ::: "memory");
}

// ---------------------------------------------------------------------------
// Templated implicit-GEMM conv1d on WMMA.
//   IN  : LDS, time-major [rows][CinP] f16, interior starts at row LT
//   OUT : LDS, time-major [rows][CoutP] f16 (interior at row OutLT), optional
//   OUTF: LDS, channel-major [Cout][Tout] f32, optional (for mean)
//   SKIP: LDS, same layout as OUT, optional
//   W2  : global f16, A-operand-swizzled [Kw][Cout][CinP] (32-k chunks, contiguous k order)
// Per lane:  B = one 32B contiguous LDS load, A = one 32B contiguous load,
//            D = one 16B packed LDS store.  No conditionals in the MAC loop.
// ---------------------------------------------------------------------------
template<int CinP, int Cout, int Tout, int CoutP, int Kw, int Stride,
         bool Relu, bool HasSkip, bool F32Out, int OutLT>
__device__ __forceinline__ void conv_t(
    const _Float16* __restrict__ IN, _Float16* __restrict__ OUT,
    float* __restrict__ OUTF, const _Float16* __restrict__ SKIP,
    const _Float16* __restrict__ W2, const float* __restrict__ Bias, int tid)
{
  constexpr int PAD    = (Kw - 1) / 2;
  constexpr int CHUNKS = CinP / 32;
  constexpr int TILEST = (Tout + 15) / 16;
  constexpr int TILES  = (Cout / 16) * TILEST;

  const int lane = tid & 31;
  const int wave = tid >> 5;
  const int m    = lane & 15;
  const int half = lane >> 4;

  for (int tile = wave; tile < TILES; tile += 8) {
    const int tc  = tile / TILEST;
    const int tt  = tile - tc * TILEST;
    const int co0 = tc << 4;
    const int t   = (tt << 4) + m;
    const int row0 = t * Stride - PAD + LT;

    v8f acc = {};
    const _Float16* arow0 = W2 + (((co0 + m) * CHUNKS) << 5) + (half << 4);
    const _Float16* brow0 = IN + row0 * CinP + (half << 4);

#pragma unroll
    for (int kk = 0; kk < Kw; ++kk) {
      const _Float16* arow = arow0 + ((kk * Cout * CHUNKS) << 5);
      const _Float16* brow = brow0 + kk * CinP;
#pragma unroll
      for (int c = 0; c < CHUNKS; ++c) {
        v16h a  = *(const v16h*)(arow + (c << 5));  // swizzled A, contiguous 32B
        v16h bv = *(const v16h*)(brow + (c << 5));  // time-major B, contiguous 32B
        acc = __builtin_amdgcn_wmma_f32_16x16x32_f16(
            false, a, false, bv, (short)0, acc, false, false);
      }
    }

    if (t < Tout) {
      const int cob = co0 + (half << 3);   // 8 consecutive output channels
      float vals[8];
#pragma unroll
      for (int v = 0; v < 8; ++v) vals[v] = acc[v] + Bias[cob + v];
      if (HasSkip) {
        v8h sk = *(const v8h*)(SKIP + (OutLT + t) * CoutP + cob);
#pragma unroll
        for (int v = 0; v < 8; ++v) vals[v] += (float)sk[v];
      }
      if (Relu) {
#pragma unroll
        for (int v = 0; v < 8; ++v) vals[v] = fmaxf(vals[v], 0.f);
      }
      if (OUT) {
        v8h o;
#pragma unroll
        for (int v = 0; v < 8; ++v) o[v] = (_Float16)vals[v];
        *(v8h*)(OUT + (OutLT + t) * CoutP + cob) = o;
      }
      if (F32Out) {
#pragma unroll
        for (int v = 0; v < 8; ++v) OUTF[(cob + v) * Tout + t] = vals[v];
      }
    }
  }
}

// ---------------------------------------------------------------------------
// Weight reorg: W[co][ci][kk] f32 -> W2[kk][co][chunk][pos] f16, zero-padded,
// pos = K (contiguous A-operand lane order). grid = 16 layers.
// ---------------------------------------------------------------------------
struct RP {
  const float* src[16];
  int off[16];
  int cout[16], cin[16], cinp[16], kw[16];
};

__global__ void k_reorg(RP R, _Float16* __restrict__ W2) {
  const int l = blockIdx.x;
  const float* __restrict__ src = R.src[l];
  const int Cout = R.cout[l], Cin = R.cin[l], CinP = R.cinp[l], Kw = R.kw[l];
  const int shC = (CinP == 32) ? 5 : ((CinP == 64) ? 6 : 7);
  const int shO = (Cout == 64) ? 6 : 7;
  const int n = Kw * Cout * CinP;
  _Float16* dst = W2 + R.off[l];
  for (int idx = threadIdx.x; idx < n; idx += blockDim.x) {
    const int ci  = idx & (CinP - 1);
    const int rem = idx >> shC;
    const int co  = rem & (Cout - 1);
    const int kk  = rem >> shO;
    const float v = (ci < Cin) ? src[(co * Cin + ci) * Kw + kk] : 0.f;
    const int K   = ci & 31;
    const int pos = K;
    dst[idx - K + pos] = (_Float16)v;
  }
}

// ---------------------------------------------------------------------------
// Kernel 1: PRNG selection + gather + stem conv + maxpool. grid=32, block=256.
// ---------------------------------------------------------------------------
#define K1_IN_H   0
#define K1_SO_H   (IN_ROWS * IN_CP)                     // 33024 halves
#define K1_EG_B   ((K1_SO_H + T_STEM * 64) * 2)         // 130048
#define K1_SEL_B  (K1_EG_B + 100 * 4)                   // 130448
#define K1_TRI_B  (K1_SEL_B + NSEL * 4)                 // 130548
#define K1_SMEM   (K1_TRI_B + NSEL * 4 + 40)            // 130688

__global__ void k_gather_stem(const float* __restrict__ x,
                              const _Float16* __restrict__ W2stem,
                              const float* __restrict__ stem_b,
                              _Float16* __restrict__ pooled) {
  extern __shared__ __align__(128) char smem[];
  _Float16* IN  = (_Float16*)smem;              // [1032][32]
  _Float16* SO  = IN + K1_SO_H;                 // [500][64] (no halo)
  float* eg     = (float*)(smem + K1_EG_B);
  int*   sel    = (int*)(smem + K1_SEL_B);
  int*   trial  = (int*)(smem + K1_TRI_B);
  const int tid = threadIdx.x;
  const int b   = blockIdx.x;

  // k1,k2 = split(key(42)): cipher blocks (0,2),(1,3) under key (0,42)
  unsigned A0, A1, B0, B1;
  tf2x32(0u, 42u, 0u, 2u, A0, A1);
  tf2x32(0u, 42u, 1u, 3u, B0, B1);

  // zero the stem input buffer (halo rows + pad channels) while making eg
  {
    const int4 z = make_int4(0, 0, 0, 0);
    int4* zp = (int4*)smem;
    for (int i = tid; i < (IN_ROWS * IN_CP * 2) / 16; i += blockDim.x) zp[i] = z;
  }
  if (tid < NE) eg[tid] = uniform_at(A0, B0, b * NE + tid, BATCH * NE);
  __syncthreads();

  if (tid == 0) {  // top_k(eg, 25) descending, first index wins ties
    for (int j = 0; j < NSEL; ++j) {
      float best = -3.f; int bi = 0;
      for (int e = 0; e < NE; ++e) {
        const float v = eg[e];
        if (v > best) { best = v; bi = e; }
      }
      sel[j] = bi;
      eg[bi] = -4.f;
    }
  }
  __syncthreads();

  if (tid < NSEL) {  // argmax over trials of uniform(k2, (B,25,TR))
    float best = -3.f; int bi = 0;
    for (int tr = 0; tr < NTR; ++tr) {
      const float v = uniform_at(A1, B1, (b * NSEL + tid) * NTR + tr,
                                 BATCH * NSEL * NTR);
      if (v > best) { best = v; bi = tr; }
    }
    trial[tid] = bi;
  }
  __syncthreads();

  // gather: IN[(LT + t)][j] = x[b, sel[j], trial[j], t]   (time-major)
  for (int idx = tid; idx < NSEL * NT; idx += blockDim.x) {
    const int j = idx / NT;
    const int t = idx - j * NT;
    const size_t off = (((size_t)b * NE + sel[j]) * NTR + trial[j]) * NT + t;
    IN[(LT + t) * IN_CP + j] = (_Float16)x[off];
  }
  __syncthreads();

  // stem conv 25(->32)->64, k7, s2, relu; output [500][64], no halo
  conv_t<IN_CP, 64, T_STEM, 64, 7, 2, true, false, false, 0>(
      IN, SO, nullptr, nullptr, W2stem, stem_b, tid);
  __syncthreads();

  // maxpool k3 s2 p1 -> pooled[b] time-major [264][64] with zero halo rows
  _Float16* outp = pooled + (size_t)b * (P_ROWS * P_CP);
  for (int g = tid; g < P_ROWS * 8; g += blockDim.x) {
    const int row = g >> 3;
    const int cg  = (g & 7) << 3;
    const int t   = row - LT;
    v8h o;
    if (t >= 0 && t < T_POOL) {
      float mx[8];
#pragma unroll
      for (int e = 0; e < 8; ++e) mx[e] = -3.0e38f;
#pragma unroll
      for (int d = 0; d < 3; ++d) {
        const int s = 2 * t - 1 + d;
        if (s >= 0 && s < T_STEM) {
          v8h sv = *(const v8h*)(SO + s * 64 + cg);
#pragma unroll
          for (int e = 0; e < 8; ++e) mx[e] = fmaxf(mx[e], (float)sv[e]);
        }
      }
#pragma unroll
      for (int e = 0; e < 8; ++e) o[e] = (_Float16)mx[e];
    } else {
#pragma unroll
      for (int e = 0; e < 8; ++e) o[e] = (_Float16)0.f;
    }
    *(v8h*)(outp + row * P_CP + cg) = o;
  }
}

// ---------------------------------------------------------------------------
// Kernel 2: one (batch, branch) residual tower per block. grid=(3,32), 256 thr.
// LDS: X/Y1/Y2 [264][64] f16, Z1/SC [136][128] f16, ZF [128][125] f32 = 235008B
// ---------------------------------------------------------------------------
#define K2_XH   0
#define K2_Y1H  (P_ROWS * P_CP)            // 16896
#define K2_Y2H  (K2_Y1H * 2)               // 33792
#define K2_Z1H  (K2_Y1H * 3)               // 50688
#define K2_SCH  (K2_Z1H + D_ROWS * D_CP)   // 68096
#define K2_F16H (K2_SCH + D_ROWS * D_CP)   // 85504 halves
#define K2_ZFB  (K2_F16H * 2)              // 171008 bytes
#define K2_SMEM (K2_ZFB + 128 * T_DOWN * 4)  // 235008 bytes

struct BrP {
  int woff[3][5];               // 0:b1c1 1:b1c2 2:b2c1 3:down 4:b2c2
  const float* bias[3][5];
};

template<int KW>
__device__ __forceinline__ void branch_body(
    _Float16* X, _Float16* Y1, _Float16* Y2, _Float16* Z1, _Float16* SC,
    float* ZF, const _Float16* __restrict__ W2, const int* wo,
    const float* const* bi, int tid)
{
  conv_t<64, 64, T_POOL, 64, KW, 1, true, false, false, LT>(
      X, Y1, nullptr, nullptr, W2 + wo[0], bi[0], tid);
  __syncthreads();
  conv_t<64, 64, T_POOL, 64, KW, 1, true, true, false, LT>(
      Y1, Y2, nullptr, X, W2 + wo[1], bi[1], tid);
  __syncthreads();
  conv_t<64, 128, T_DOWN, 128, KW, 2, true, false, false, LT>(
      Y2, Z1, nullptr, nullptr, W2 + wo[2], bi[2], tid);
  __syncthreads();
  conv_t<64, 128, T_DOWN, 128, 1, 2, false, false, false, LT>(
      Y2, SC, nullptr, nullptr, W2 + wo[3], bi[3], tid);
  __syncthreads();
  conv_t<128, 128, T_DOWN, 128, KW, 1, true, true, true, LT>(
      Z1, nullptr, ZF, SC, W2 + wo[4], bi[4], tid);
}

__global__ void k_branch(const _Float16* __restrict__ pooled,
                         float* __restrict__ feats,
                         const _Float16* __restrict__ W2, BrP P) {
  extern __shared__ __align__(128) char smem[];
  _Float16* X  = (_Float16*)smem + K2_XH;
  _Float16* Y1 = (_Float16*)smem + K2_Y1H;
  _Float16* Y2 = (_Float16*)smem + K2_Y2H;
  _Float16* Z1 = (_Float16*)smem + K2_Z1H;
  _Float16* SC = (_Float16*)smem + K2_SCH;
  float*    ZF = (float*)(smem + K2_ZFB);

  const int br  = blockIdx.x;
  const int b   = blockIdx.y;
  const int tid = threadIdx.x;

  // stage pooled[b] (33792 B) into X via async global->LDS (ASYNCcnt path)
  {
    const _Float16* src = pooled + (size_t)b * (P_ROWS * P_CP);
    const unsigned xlds = (unsigned)(size_t)(void*)X;
    for (int i = tid; i < (P_ROWS * P_CP * 2) / 16; i += blockDim.x)
      async_ld_b128(xlds + (unsigned)(i * 16), src, i * 16);
  }
  // zero Y1..SC halos while the async copy is in flight
  {
    const int4 z = make_int4(0, 0, 0, 0);
    int4* zp = (int4*)(smem + K2_Y1H * 2);
    for (int i = tid; i < (K2_F16H - K2_Y1H) * 2 / 16; i += blockDim.x) zp[i] = z;
  }
  wait_async0();
  __syncthreads();

  const int* wo = P.woff[br];
  const float* const* bi = P.bias[br];
  if (br == 0)      branch_body<3>(X, Y1, Y2, Z1, SC, ZF, W2, wo, bi, tid);
  else if (br == 1) branch_body<5>(X, Y1, Y2, Z1, SC, ZF, W2, wo, bi, tid);
  else              branch_body<7>(X, Y1, Y2, Z1, SC, ZF, W2, wo, bi, tid);
  __syncthreads();

  for (int c = tid; c < 128; c += blockDim.x) {  // temporal mean
    float s = 0.f;
    for (int t = 0; t < T_DOWN; ++t) s += ZF[c * T_DOWN + t];
    feats[((size_t)b * 3 + br) * 128 + c] = s * (1.0f / (float)T_DOWN);
  }
}

// ---------------------------------------------------------------------------
// Kernel 3: FC head. grid=32, block=128.
// ---------------------------------------------------------------------------
__global__ void k_head(const float* __restrict__ feats,
                       const float* __restrict__ fc_w,  const float* __restrict__ fc_b,
                       const float* __restrict__ cl1_w, const float* __restrict__ cl1_b,
                       const float* __restrict__ cl2_w, const float* __restrict__ cl2_b,
                       float* __restrict__ out) {
  __shared__ float emb[128];
  __shared__ float hc[128];
  const int b   = blockIdx.x;
  const int tid = threadIdx.x;
  const float* f = feats + (size_t)b * 384;

  float s = fc_b[tid];
  for (int i = 0; i < 384; ++i) s += f[i] * fc_w[i * 128 + tid];
  emb[tid] = s;
  __syncthreads();

  float h = cl1_b[tid];
  for (int i = 0; i < 128; ++i) h += emb[i] * cl1_w[i * 128 + tid];
  hc[tid] = fmaxf(h, 0.f);
  __syncthreads();

  if (tid < 2) {
    float o = cl2_b[tid];
    for (int i = 0; i < 128; ++i) o += hc[i] * cl2_w[i * 2 + tid];
    out[b * 2 + tid] = o;
  }
}

// ---------------------------------------------------------------------------
// Launch. d_in order: x, padding_mask, params leaves (sorted-key tree order):
// branches[0..2]{b1:{b1,b2,w1,w2}, b2:{b1,b2,bd,w1,w2,wd}}, cl1_b, cl1_w,
// cl2_b, cl2_w, fc_b, fc_w, stem_b, stem_w.  (40 inputs total)
// ---------------------------------------------------------------------------
extern "C" void kernel_launch(void* const* d_in, const int* in_sizes, int n_in,
                              void* d_out, int out_size, void* d_ws, size_t ws_size,
                              hipStream_t stream) {
  (void)in_sizes; (void)out_size; (void)ws_size;
  if (n_in < 40) return;
  auto f = [&](int i) { return (const float*)d_in[i]; };
  const float* x = f(0);

  // workspace layout (bytes)
  _Float16* W2     = (_Float16*)d_ws;                          // 1,060,864 B
  _Float16* pooled = (_Float16*)((char*)d_ws + 1060864);       // 1,081,344 B
  float*    feats  = (float*)((char*)d_ws + 2142208);          //    49,152 B

  RP rp; BrP bp;
  int off = 0, l = 0;
  auto addL = [&](const float* s, int Co, int Ci, int Cp, int Kw_) {
    rp.src[l] = s; rp.off[l] = off;
    rp.cout[l] = Co; rp.cin[l] = Ci; rp.cinp[l] = Cp; rp.kw[l] = Kw_;
    const int r = off; off += Kw_ * Co * Cp; ++l; return r;
  };
  addL(f(39), 64, 25, 32, 7);                       // stem at offset 0
  for (int br = 0; br < 3; ++br) {
    const int base = 2 + br * 10;
    const int k = 3 + 2 * br;
    bp.woff[br][0] = addL(f(base + 2), 64, 64, 64, k);     // b1.w1
    bp.woff[br][1] = addL(f(base + 3), 64, 64, 64, k);     // b1.w2
    bp.woff[br][2] = addL(f(base + 7), 128, 64, 64, k);    // b2.w1
    bp.woff[br][3] = addL(f(base + 9), 128, 64, 64, 1);    // b2.wd
    bp.woff[br][4] = addL(f(base + 8), 128, 128, 128, k);  // b2.w2
    bp.bias[br][0] = f(base + 0);   // b1.b1
    bp.bias[br][1] = f(base + 1);   // b1.b2
    bp.bias[br][2] = f(base + 4);   // b2.b1
    bp.bias[br][3] = f(base + 6);   // b2.bd
    bp.bias[br][4] = f(base + 5);   // b2.b2
  }

  k_reorg<<<16, 256, 0, stream>>>(rp, W2);
  k_gather_stem<<<BATCH, 256, K1_SMEM, stream>>>(x, W2, f(38), pooled);
  dim3 g2(3, BATCH);
  k_branch<<<g2, 256, K2_SMEM, stream>>>(pooled, feats, W2, bp);
  k_head<<<BATCH, 128, 0, stream>>>(feats, f(37), f(36), f(33), f(32),
                                    f(35), f(34), (float*)d_out);
}
